// Net_cond_one_49830210568745
// MI455X (gfx1250) — matrix operand order repaired and checked by reading, not compiled, not run on hardware
//
#include <hip/hip_runtime.h>
#include <stdint.h>

// ---------------------------------------------------------------------------
// GNS-style GNN forward pass for MI455X (gfx1250), bf16 WMMA + f32 accumulate.
// Weights are pre-converted to WMMA fragment-major bf16 so each v_wmma's B
// operand is fed by two coalesced global_load_b128 (L2-resident), and the
// GEMM K-loops over LDS-resident A need no barriers.
// ---------------------------------------------------------------------------

typedef __attribute__((ext_vector_type(16))) __bf16 v16bf;
typedef __attribute__((ext_vector_type(8)))  float  v8f;

static constexpr int   N_PART = 30000;
static constexpr int   N_EDGE = 300000;
static constexpr int   NLAYER = 10;
static constexpr int   TWIN   = 5;
static constexpr float RADIUS = 0.05f;

union AFrag { v16bf v; uint32_t u[8]; };

__device__ __forceinline__ uint16_t f2bf(float f) {
  uint32_t u = __float_as_uint(f);
  uint32_t r = u + 0x7FFFu + ((u >> 16) & 1u);   // round-to-nearest-even
  return (uint16_t)(r >> 16);
}
__device__ __forceinline__ float bf2f(uint16_t h) {
  return __uint_as_float(((uint32_t)h) << 16);
}

__device__ __forceinline__ void zero_acc(v8f acc[8]) {
#pragma unroll
  for (int t = 0; t < 8; ++t)
#pragma unroll
    for (int i = 0; i < 8; ++i) acc[t][i] = 0.f;
}

// A fragment (16x32 bf16): lane m = lane&15, half h = lane>>4.
// elements 0..7  <-> K = colb + 16h + 0..7   (16 contiguous bytes)
// elements 8..15 <-> K = colb + 16h + 8 + 0..7
__device__ __forceinline__ void load_afrag(const uint16_t* s, int stride, int row0,
                                           int colb, AFrag& a) {
  int lane = threadIdx.x & 31;
  int m = lane & 15, h = lane >> 4;
  const uint32_t* p0 = (const uint32_t*)(s + (size_t)(row0 + m) * stride + colb + 16 * h);
  const uint32_t* p1 = (const uint32_t*)(s + (size_t)(row0 + m) * stride + colb + 16 * h + 8);
#pragma unroll
  for (int i = 0; i < 4; ++i) { a.u[i] = p0[i]; a.u[4 + i] = p1[i]; }
}

// B operand direct from global fragment-major weights:
// Wf[((kc*8 + ntile)*32 + lane)*16 + j], element j <-> K = kc*32 + 16h + j,
// col = ntile*16 + (lane&15).  32 contiguous bytes per lane -> 2x b128.
__device__ __forceinline__ void wmma8_g(const AFrag& a, const uint16_t* Wf, int kc,
                                        v8f acc[8]) {
  int lane = threadIdx.x & 31;
#pragma unroll
  for (int t = 0; t < 8; ++t) {
    AFrag b;
    const uint32_t* p = (const uint32_t*)(Wf + ((size_t)(kc * 8 + t) * 32 + lane) * 16);
#pragma unroll
    for (int i = 0; i < 8; ++i) b.u[i] = p[i];
    acc[t] = __builtin_amdgcn_wmma_f32_16x16x32_bf16(false, a.v, false, b.v,
                                                     (short)0, acc[t], false, false);
  }
}

// GEMM with A resident in LDS; barrier-free K-loop (B comes from global).
__device__ __forceinline__ void gemm_lds_g(const uint16_t* sA, int strideA,
                                           const uint16_t* Wf, int K, v8f acc[8]) {
  int warp = threadIdx.x >> 5;
  for (int kc = 0; kc < (K >> 5); ++kc) {
    AFrag a;
    load_afrag(sA, strideA, warp * 16, kc * 32, a);
    wmma8_g(a, Wf, kc, acc);
  }
}

// D fragment: row m = 8*h + i (i = vgpr index), col n = 16*t + (lane&15)
__device__ __forceinline__ void store_relu_bf16(const v8f acc[8], const float* bias,
                                                uint16_t* s_out, int stride) {
  int lane = threadIdx.x & 31, warp = threadIdx.x >> 5;
  int n = lane & 15, h = lane >> 4;
#pragma unroll
  for (int t = 0; t < 8; ++t) {
    float bv = bias[t * 16 + n];
#pragma unroll
    for (int i = 0; i < 8; ++i) {
      float v = acc[t][i] + bv;
      v = v > 0.f ? v : 0.f;
      s_out[(size_t)(warp * 16 + 8 * h + i) * stride + t * 16 + n] = f2bf(v);
    }
  }
}

// Gathered A staging: row = concat of up to three bf16 segments along K,
// with optional row-index indirection per segment (the GNN concat fusion).
__device__ __forceinline__ void stage_a_seg(int r0, int M, int kc,
                                            const uint16_t* seg0, int w0,
                                            const uint16_t* seg1, int w1, const int* idx1,
                                            const uint16_t* seg2, int w2, const int* idx2,
                                            uint16_t* s_a) {
#pragma unroll
  for (int it = 0; it < 16; ++it) {
    int e = threadIdx.x + it * 256;
    int r = e >> 5, c = e & 31;
    int gr = r0 + r;
    uint16_t v = 0;
    if (gr < M) {
      int k = kc * 32 + c;
      if (k < w0) {
        v = seg0[(size_t)gr * w0 + k];
      } else if (k < w0 + w1) {
        int rr = idx1 ? idx1[gr] : gr;
        v = seg1[(size_t)rr * w1 + (k - w0)];
      } else {
        int rr = idx2 ? idx2[gr] : gr;
        v = seg2[(size_t)rr * w2 + (k - w0 - w1)];
      }
    }
    s_a[r * 40 + c] = v;   // stride 40 bf16 (80B) keeps 16B chunks aligned
  }
}

// ---------------------------------------------------------------------------
// Fused:  LN( relu(relu(A@W1+b1)@W2+b2)@W3 + b3 ) * g + be  [+ residual]
// A = concat(seg0, seg1[idx1], seg2[idx2]); 128-row tile per block; 8 waves.
// Double-buffered A staging: one barrier per K-chunk in GEMM1, none in 2/3.
// ---------------------------------------------------------------------------
__global__ __launch_bounds__(256) void k_mlp3(
    int M, int K,
    const uint16_t* seg0, int w0,
    const uint16_t* seg1, int w1, const int* idx1,
    const uint16_t* seg2, int w2, const int* idx2,
    const uint16_t* W1, const float* b1,
    const uint16_t* W2, const float* b2,
    const uint16_t* W3, const float* b3,
    const float* gamma, const float* beta,
    const float* resid, float* out_f32, uint16_t* out_bf16)
{
  __shared__ alignas(16) char smem[92160];
  uint16_t* s_a0 = (uint16_t*)smem;                  // 10240 B (128 x 40 bf16)
  uint16_t* s_a1 = (uint16_t*)(smem + 10240);        // 10240 B
  uint16_t* s_h1 = (uint16_t*)(smem + 20480);        // 34816 B (128 x 136 bf16)
  uint16_t* s_h2 = (uint16_t*)(smem + 55296);        // 34816 B -> 90112
  float*    s_f  = (float*)smem;                     // 67584 B overlay (dead bufs)
  float*    s_red= (float*)(smem + 90112);           //  2048 B

  int r0 = blockIdx.x * 128;
  int warp = threadIdx.x >> 5;
  v8f acc[8];

  // ---- GEMM1: gathered A (K = w0+w1+w2), double-buffered staging ----
  int KC = K >> 5;
  zero_acc(acc);
  stage_a_seg(r0, M, 0, seg0, w0, seg1, w1, idx1, seg2, w2, idx2, s_a0);
  __syncthreads();
  for (int kc = 0; kc < KC; ++kc) {
    uint16_t* cur = (kc & 1) ? s_a1 : s_a0;
    uint16_t* nxt = (kc & 1) ? s_a0 : s_a1;
    if (kc + 1 < KC)
      stage_a_seg(r0, M, kc + 1, seg0, w0, seg1, w1, idx1, seg2, w2, idx2, nxt);
    AFrag a;
    load_afrag(cur, 40, warp * 16, 0, a);
    wmma8_g(a, W1, kc, acc);
    __syncthreads();
  }
  store_relu_bf16(acc, b1, s_h1, 136);
  __syncthreads();

  // ---- GEMM2 (barrier-free K-loop) ----
  zero_acc(acc);
  gemm_lds_g(s_h1, 136, W2, 128, acc);
  store_relu_bf16(acc, b2, s_h2, 136);
  __syncthreads();

  // ---- GEMM3 (no relu) ----
  zero_acc(acc);
  gemm_lds_g(s_h2, 136, W3, 128, acc);
  __syncthreads();        // all waves done reading s_h2 before s_f overlays it
  {
    int lane = threadIdx.x & 31;
    int n = lane & 15, h = lane >> 4;
#pragma unroll
    for (int t = 0; t < 8; ++t) {
      float bv = b3[t * 16 + n];
#pragma unroll
      for (int i = 0; i < 8; ++i)
        s_f[(size_t)(warp * 16 + 8 * h + i) * 132 + t * 16 + n] = acc[t][i] + bv;
    }
  }
  __syncthreads();

  // ---- LayerNorm (+ residual), dual f32/bf16 output ----
  int r = threadIdx.x >> 1, hf = threadIdx.x & 1;
  const float* rowp = s_f + (size_t)r * 132 + hf * 64;
  float s = 0.f, q = 0.f;
  for (int j = 0; j < 64; ++j) { float v = rowp[j]; s += v; q += v * v; }
  s_red[r * 2 + hf] = s;
  s_red[256 + r * 2 + hf] = q;
  __syncthreads();
  float mean = (s_red[r * 2] + s_red[r * 2 + 1]) * (1.f / 128.f);
  float var  = (s_red[256 + r * 2] + s_red[256 + r * 2 + 1]) * (1.f / 128.f) - mean * mean;
  float inv = rsqrtf(var + 1e-5f);
  int gr = r0 + r;
  if (gr < M) {
    for (int j = 0; j < 64; ++j) {
      int col = hf * 64 + j;
      float v = (s_f[(size_t)r * 132 + col] - mean) * inv * gamma[col] + beta[col];
      if (resid) v += resid[(size_t)gr * 128 + col];
      out_f32[(size_t)gr * 128 + col] = v;
      out_bf16[(size_t)gr * 128 + col] = f2bf(v);
    }
  }
}

// ---------------------------------------------------------------------------
// Decoder: relu(nodes@W1+b1) -> relu(@W2+b2) -> @W3(128x2)+b3 + Euler step.
// ---------------------------------------------------------------------------
__global__ __launch_bounds__(256) void k_decoder(
    const uint16_t* nodesb,
    const uint16_t* W1, const float* b1,
    const uint16_t* W2, const float* b2,
    const float* W3, const float* b3,
    const float* pos, const float* vel,
    const int* nonk, const float* tgt,
    float* out)
{
  __shared__ alignas(16) char smem[90112];
  uint16_t* s_a0 = (uint16_t*)smem;
  uint16_t* s_a1 = (uint16_t*)(smem + 10240);
  uint16_t* s_h1 = (uint16_t*)(smem + 20480);
  uint16_t* s_h2 = (uint16_t*)(smem + 55296);

  int r0 = blockIdx.x * 128;
  int warp = threadIdx.x >> 5;
  v8f acc[8];

  zero_acc(acc);
  stage_a_seg(r0, N_PART, 0, nodesb, 128, nullptr, 0, nullptr, nullptr, 0, nullptr, s_a0);
  __syncthreads();
  for (int kc = 0; kc < 4; ++kc) {
    uint16_t* cur = (kc & 1) ? s_a1 : s_a0;
    uint16_t* nxt = (kc & 1) ? s_a0 : s_a1;
    if (kc + 1 < 4)
      stage_a_seg(r0, N_PART, kc + 1, nodesb, 128, nullptr, 0, nullptr,
                  nullptr, 0, nullptr, nxt);
    AFrag a;
    load_afrag(cur, 40, warp * 16, 0, a);
    wmma8_g(a, W1, kc, acc);
    __syncthreads();
  }
  store_relu_bf16(acc, b1, s_h1, 136);
  __syncthreads();

  zero_acc(acc);
  gemm_lds_g(s_h1, 136, W2, 128, acc);
  store_relu_bf16(acc, b2, s_h2, 136);
  __syncthreads();

  int r = threadIdx.x >> 1, c = threadIdx.x & 1;
  int gr = r0 + r;
  if (gr < N_PART) {
    float d = b3[c];
    const uint16_t* hrow = s_h2 + (size_t)r * 136;
    for (int k = 0; k < 128; ++k) d += bf2f(hrow[k]) * W3[k * 2 + c];
    float pa = d;
    float pp = pos[(size_t)gr * 2 + c] + vel[(size_t)gr * (TWIN * 2) + (TWIN - 1) * 2 + c] + pa;
    if (nonk[gr] == 0) pp = tgt[(size_t)gr * 2 + c];
    out[(size_t)gr * 2 + c] = pa;                              // pred_acc
    out[(size_t)N_PART * 2 + (size_t)gr * 2 + c] = pp;         // pred_pos
  }
}

// ---------------------------------------------------------------------------
// Small kernels: feature construction, weight conversion, segment-sum.
// ---------------------------------------------------------------------------
__global__ void k_node_attr(const float* vel, const float* pos, const float* bounds,
                            const int* ptype, const float* emb, uint16_t* out)
{
  int i = blockIdx.x * 256 + threadIdx.x;
  if (i >= N_PART) return;
  uint16_t* o = out + (size_t)i * 32;
#pragma unroll
  for (int j = 0; j < 10; ++j) o[j] = f2bf(vel[(size_t)i * 10 + j]);
  float px = pos[(size_t)i * 2], py = pos[(size_t)i * 2 + 1];
  float lo0 = bounds[0], hi0 = bounds[1], lo1 = bounds[2], hi1 = bounds[3];
  float inv = 1.f / RADIUS;
  float d0 = (px - lo0) * inv, d1 = (py - lo1) * inv;
  float d2 = (hi0 - px) * inv, d3 = (hi1 - py) * inv;
  o[10] = f2bf(fminf(1.f, fmaxf(-1.f, d0)));
  o[11] = f2bf(fminf(1.f, fmaxf(-1.f, d1)));
  o[12] = f2bf(fminf(1.f, fmaxf(-1.f, d2)));
  o[13] = f2bf(fminf(1.f, fmaxf(-1.f, d3)));
  int tp = ptype[i];
#pragma unroll
  for (int j = 0; j < 16; ++j) o[14 + j] = f2bf(emb[(size_t)tp * 16 + j]);
  o[30] = 0; o[31] = 0;
}

__global__ void k_edge_attr(const float* pos, const int* snd, const int* rcv, uint16_t* out)
{
  int i = blockIdx.x * 256 + threadIdx.x;
  if (i >= N_EDGE) return;
  int s = snd[i], r = rcv[i];
  float inv = 1.f / RADIUS;
  float dx = (pos[(size_t)s * 2]     - pos[(size_t)r * 2])     * inv;
  float dy = (pos[(size_t)s * 2 + 1] - pos[(size_t)r * 2 + 1]) * inv;
  float dn = sqrtf(dx * dx + dy * dy);
  uint16_t* o = out + (size_t)i * 32;
  o[0] = f2bf(dx); o[1] = f2bf(dy); o[2] = f2bf(dn);
#pragma unroll
  for (int j = 3; j < 32; ++j) o[j] = 0;
}

// Convert W (src_rows x 128 f32, zero-pad K to Kpad) into WMMA fragment-major
// bf16: dst[((kc*8+ntile)*32+lane)*16 + j], j <-> K=kc*32+16h+j, col=16*ntile+n.
__global__ void k_cvt_wfrag(const float* src, uint16_t* dst, int src_rows, int Kpad)
{
  int i = blockIdx.x * 256 + threadIdx.x;
  if (i >= Kpad * 128) return;
  int j     = i & 15;
  int lane  = (i >> 4) & 31;
  int ntile = (i >> 9) & 7;
  int kc    = i >> 12;
  int n = lane & 15, h = lane >> 4;
  int Krow = kc * 32 + 16 * h + j;
  int col  = ntile * 16 + n;
  float v = (Krow < src_rows) ? src[(size_t)Krow * 128 + col] : 0.f;
  dst[i] = f2bf(v);
}

__global__ void k_zero(float* p, int n)
{
  int i = blockIdx.x * 256 + threadIdx.x;
  if (i < n) p[i] = 0.f;
}

__global__ void k_scatter(const float* edges, const int* rcv, float* agg)
{
  int i = blockIdx.x * 256 + threadIdx.x;
  if (i >= N_EDGE * 128) return;
  int e = i >> 7, c = i & 127;
  float v = edges[(size_t)e * 128 + c];
  unsafeAtomicAdd(&agg[(size_t)rcv[e] * 128 + c], v);   // native global_atomic_add_f32
}

__global__ void k_f32_to_bf16(const float* src, uint16_t* dst, int n)
{
  int i = blockIdx.x * 256 + threadIdx.x;
  if (i < n) dst[i] = f2bf(src[i]);
}

// ---------------------------------------------------------------------------
extern "C" void kernel_launch(void* const* d_in, const int* in_sizes, int n_in,
                              void* d_out, int out_size, void* d_ws, size_t ws_size,
                              hipStream_t stream)
{
  (void)in_sizes; (void)n_in; (void)out_size; (void)ws_size;

  const float* vel    = (const float*)d_in[0];
  const float* pos    = (const float*)d_in[1];
  const float* bounds = (const float*)d_in[2];
  const float* tgt    = (const float*)d_in[3];
  const int*   ptype  = (const int*)d_in[4];
  const int*   nonk   = (const int*)d_in[5];
  const int*   snd    = (const int*)d_in[6];
  const int*   rcv    = (const int*)d_in[7];
  const float* emb    = (const float*)d_in[8];
  const float *neW1 = (const float*)d_in[9],  *neb1 = (const float*)d_in[10];
  const float *neW2 = (const float*)d_in[11], *neb2 = (const float*)d_in[12];
  const float *neW3 = (const float*)d_in[13], *neb3 = (const float*)d_in[14];
  const float *neg  = (const float*)d_in[15], *nebe = (const float*)d_in[16];
  const float *eeW1 = (const float*)d_in[17], *eeb1 = (const float*)d_in[18];
  const float *eeW2 = (const float*)d_in[19], *eeb2 = (const float*)d_in[20];
  const float *eeW3 = (const float*)d_in[21], *eeb3 = (const float*)d_in[22];
  const float *eeg  = (const float*)d_in[23], *eebe = (const float*)d_in[24];
  const float *geW1 = (const float*)d_in[25], *geb1 = (const float*)d_in[26];
  const float *geW2 = (const float*)d_in[27], *geb2 = (const float*)d_in[28];
  const float *geW3 = (const float*)d_in[29], *geb3 = (const float*)d_in[30];
  const float *geg  = (const float*)d_in[31], *gebe = (const float*)d_in[32];
  const float *gnW1 = (const float*)d_in[33], *gnb1 = (const float*)d_in[34];
  const float *gnW2 = (const float*)d_in[35], *gnb2 = (const float*)d_in[36];
  const float *gnW3 = (const float*)d_in[37], *gnb3 = (const float*)d_in[38];
  const float *gng  = (const float*)d_in[39], *gnbe = (const float*)d_in[40];
  const float *deW1 = (const float*)d_in[41], *deb1 = (const float*)d_in[42];
  const float *deW2 = (const float*)d_in[43], *deb2 = (const float*)d_in[44];
  const float *deW3 = (const float*)d_in[45], *deb3 = (const float*)d_in[46];
  float* out = (float*)d_out;

  // ---- workspace carving ----
  char* base = (char*)d_ws;
  size_t off = 0;
  auto alloc = [&](size_t bytes) -> void* {
    void* p = base + off;
    off += (bytes + 255) & ~(size_t)255;
    return p;
  };
  float*    nodes_f = (float*)alloc((size_t)N_PART * 128 * 4);
  float*    edges_f = (float*)alloc((size_t)N_EDGE * 128 * 4);
  float*    agg_f   = (float*)alloc((size_t)N_PART * 128 * 4);
  uint16_t* nodes_b = (uint16_t*)alloc((size_t)N_PART * 128 * 2);
  uint16_t* edges_b = (uint16_t*)alloc((size_t)N_EDGE * 128 * 2);
  uint16_t* agg_b   = (uint16_t*)alloc((size_t)N_PART * 128 * 2);
  uint16_t* nattr   = (uint16_t*)alloc((size_t)N_PART * 32 * 2);
  uint16_t* eattr   = (uint16_t*)alloc((size_t)N_EDGE * 32 * 2);
  uint16_t* neW1b = (uint16_t*)alloc(32 * 128 * 2);
  uint16_t* neW2b = (uint16_t*)alloc(128 * 128 * 2);
  uint16_t* neW3b = (uint16_t*)alloc(128 * 128 * 2);
  uint16_t* eeW1b = (uint16_t*)alloc(32 * 128 * 2);
  uint16_t* eeW2b = (uint16_t*)alloc(128 * 128 * 2);
  uint16_t* eeW3b = (uint16_t*)alloc(128 * 128 * 2);
  uint16_t* geW1b = (uint16_t*)alloc((size_t)NLAYER * 384 * 128 * 2);
  uint16_t* geW2b = (uint16_t*)alloc((size_t)NLAYER * 128 * 128 * 2);
  uint16_t* geW3b = (uint16_t*)alloc((size_t)NLAYER * 128 * 128 * 2);
  uint16_t* gnW1b = (uint16_t*)alloc((size_t)NLAYER * 256 * 128 * 2);
  uint16_t* gnW2b = (uint16_t*)alloc((size_t)NLAYER * 128 * 128 * 2);
  uint16_t* gnW3b = (uint16_t*)alloc((size_t)NLAYER * 128 * 128 * 2);
  uint16_t* deW1b = (uint16_t*)alloc(128 * 128 * 2);
  uint16_t* deW2b = (uint16_t*)alloc(128 * 128 * 2);

  // ---- weight conversion to fragment-major bf16 (zero K-pad for encoders) ----
  auto cvtf = [&](const float* s, uint16_t* d, int src_rows, int Kpad) {
    int total = Kpad * 128;
    k_cvt_wfrag<<<(total + 255) / 256, 256, 0, stream>>>(s, d, src_rows, Kpad);
  };
  cvtf(neW1, neW1b, 30, 32);  cvtf(neW2, neW2b, 128, 128);  cvtf(neW3, neW3b, 128, 128);
  cvtf(eeW1, eeW1b,  3, 32);  cvtf(eeW2, eeW2b, 128, 128);  cvtf(eeW3, eeW3b, 128, 128);
  for (int l = 0; l < NLAYER; ++l) {
    cvtf(geW1 + (size_t)l * 384 * 128, geW1b + (size_t)l * 384 * 128, 384, 384);
    cvtf(geW2 + (size_t)l * 128 * 128, geW2b + (size_t)l * 128 * 128, 128, 128);
    cvtf(geW3 + (size_t)l * 128 * 128, geW3b + (size_t)l * 128 * 128, 128, 128);
    cvtf(gnW1 + (size_t)l * 256 * 128, gnW1b + (size_t)l * 256 * 128, 256, 256);
    cvtf(gnW2 + (size_t)l * 128 * 128, gnW2b + (size_t)l * 128 * 128, 128, 128);
    cvtf(gnW3 + (size_t)l * 128 * 128, gnW3b + (size_t)l * 128 * 128, 128, 128);
  }
  cvtf(deW1, deW1b, 128, 128);
  cvtf(deW2, deW2b, 128, 128);

  // ---- feature construction ----
  k_node_attr<<<(N_PART + 255) / 256, 256, 0, stream>>>(vel, pos, bounds, ptype, emb, nattr);
  k_edge_attr<<<(N_EDGE + 255) / 256, 256, 0, stream>>>(pos, snd, rcv, eattr);

  int nblkN = (N_PART + 127) / 128;
  int nblkE = (N_EDGE + 127) / 128;

  // ---- encoders ----
  k_mlp3<<<nblkN, 256, 0, stream>>>(N_PART, 32,
      nattr, 32, nullptr, 0, nullptr, nullptr, 0, nullptr,
      neW1b, neb1, neW2b, neb2, neW3b, neb3, neg, nebe,
      nullptr, nodes_f, nodes_b);
  k_mlp3<<<nblkE, 256, 0, stream>>>(N_EDGE, 32,
      eattr, 32, nullptr, 0, nullptr, nullptr, 0, nullptr,
      eeW1b, eeb1, eeW2b, eeb2, eeW3b, eeb3, eeg, eebe,
      nullptr, edges_f, edges_b);

  // ---- L message-passing layers ----
  for (int l = 0; l < NLAYER; ++l) {
    // edge update: concat([edges, nodes[snd], nodes[rcv]]) fused as gather
    k_mlp3<<<nblkE, 256, 0, stream>>>(N_EDGE, 384,
        edges_b, 128, nodes_b, 128, snd, nodes_b, 128, rcv,
        geW1b + (size_t)l * 384 * 128, geb1 + l * 128,
        geW2b + (size_t)l * 128 * 128, geb2 + l * 128,
        geW3b + (size_t)l * 128 * 128, geb3 + l * 128,
        geg + l * 128, gebe + l * 128,
        edges_f, edges_f, edges_b);

    // segment_sum over receivers
    int na = N_PART * 128;
    k_zero<<<(na + 255) / 256, 256, 0, stream>>>(agg_f, na);
    k_scatter<<<(N_EDGE * 128 + 255) / 256, 256, 0, stream>>>(edges_f, rcv, agg_f);
    k_f32_to_bf16<<<(na + 255) / 256, 256, 0, stream>>>(agg_f, agg_b, na);

    // node update: concat([nodes, agg])
    k_mlp3<<<nblkN, 256, 0, stream>>>(N_PART, 256,
        nodes_b, 128, agg_b, 128, nullptr, nullptr, 0, nullptr,
        gnW1b + (size_t)l * 256 * 128, gnb1 + l * 128,
        gnW2b + (size_t)l * 128 * 128, gnb2 + l * 128,
        gnW3b + (size_t)l * 128 * 128, gnb3 + l * 128,
        gng + l * 128, gnbe + l * 128,
        nodes_f, nodes_f, nodes_b);
  }

  // ---- decoder + Euler integration ----
  k_decoder<<<nblkN, 256, 0, stream>>>(nodes_b, deW1b, deb1, deW2b, deb2, deW3, deb3,
                                       pos, vel, nonk, tgt, out);
}
